// AttentionalFactorizationMachine_3607772529177
// MI455X (gfx1250) — compile-verified, matrix operand order checked by reading, not run on hardware
//
#include <hip/hip_runtime.h>
#include <hip/hip_bf16.h>

typedef __attribute__((ext_vector_type(16))) __bf16 v16bf;
typedef __attribute__((ext_vector_type(8)))  float  v8f;

#define NFIELD 40
#define EDIM 128
#define ADIM 128
#define NPAIR 780
#define NPAIR_PAD 800
#define NTILE 25          // 800 / 32 pairs per iteration
#define BLOCK 256         // 8 waves (wave32)

__device__ __forceinline__ unsigned short f32_to_bf16(float f) {
    unsigned int u = __float_as_uint(f);
    unsigned int r = u + 0x7FFFu + ((u >> 16) & 1u);   // round-to-nearest-even
    return (unsigned short)(r >> 16);
}

__device__ __forceinline__ unsigned int pack2_bf16(float lo, float hi) {
    return (unsigned int)f32_to_bf16(lo) | ((unsigned int)f32_to_bf16(hi) << 16);
}

__global__ __launch_bounds__(BLOCK) void afm_fused_kernel(
    const float* __restrict__ x,      // [B, F, E]
    const float* __restrict__ w_w,    // [A, E]
    const float* __restrict__ w_b,    // [A]
    const float* __restrict__ h_w,    // [1, A]
    const float* __restrict__ h_b,    // [1]  (softmax shift-invariant: unused)
    const float* __restrict__ p_w,    // [1, E]
    const float* __restrict__ p_b,    // [1]
    float* __restrict__ out)          // [B]
{
    (void)h_b;
    __shared__ float xs[NFIELD * EDIM];                         // 20 KB, x row f32
    __shared__ __align__(16) unsigned short Ws[ADIM * EDIM];    // 32 KB, W bf16
    __shared__ __align__(16) unsigned short hps[2][32 * EDIM];  // 16 KB, hp tiles bf16
    __shared__ float scores[NPAIR_PAD];
    __shared__ unsigned short pairs[NPAIR_PAD];                 // (i<<8)|j
    __shared__ float afm_s[EDIM];
    __shared__ float red[BLOCK];

    const int tid  = threadIdx.x;
    const int b    = blockIdx.x;
    const int wave = tid >> 5;
    const int lane = tid & 31;

    // ---- Stage x[b] to LDS (f32), W to LDS (bf16), zero scores, build pair table
    const float4* xg = (const float4*)(x + (size_t)b * NFIELD * EDIM);
    for (int i = tid; i < NFIELD * EDIM / 4; i += BLOCK)
        ((float4*)xs)[i] = xg[i];

    const float4* wg = (const float4*)w_w;
    for (int i = tid; i < ADIM * EDIM / 4; i += BLOCK) {
        float4 v = wg[i];
        ((unsigned int*)Ws)[2*i+0] = pack2_bf16(v.x, v.y);
        ((unsigned int*)Ws)[2*i+1] = pack2_bf16(v.z, v.w);
    }
    for (int p = tid; p < NPAIR_PAD; p += BLOCK) {
        scores[p] = 0.0f;
        if (p < NPAIR) {
            int pp = p, ii = 0;
            while (pp >= NFIELD - 1 - ii) { pp -= NFIELD - 1 - ii; ii++; }
            pairs[p] = (unsigned short)((ii << 8) | (ii + 1 + pp));
        } else {
            pairs[p] = 0;
        }
    }

    // Swapped GEMM: M = a (registers), N = p (lanes), K = e.
    // C layout: reg r, lanes 0-15 -> a_local = r; lanes 16-31 -> a_local = r+8.
    const int a_base = wave * 16;
    const int ah     = (lane < 16) ? 0 : 8;   // a-half / ISA fragment K-offset
    const int ph     = (lane < 16) ? 0 : 16;  // sub-tile select for the epilogue
    const int koff   = ah;
    const int nl     = lane & 15;             // lane's p-column / W row in tile

    // Per-lane, register-indexed epilogue constants: a = a_base + r + ah
    float wbr[8], hwr[8];
    #pragma unroll
    for (int r = 0; r < 8; r++) {
        wbr[r] = w_b[a_base + r + ah];
        hwr[r] = h_w[a_base + r + ah];
    }

    __syncthreads();

    // ---- Preload A fragments (W rows, M = a): lane<16 -> a = a_base+lane
    v16bf wfrag[4];
    {
        const unsigned short* wrow = Ws + (a_base + nl) * EDIM;
        #pragma unroll
        for (int kk = 0; kk < 4; kk++) {
            const unsigned short* p0 = wrow + kk * 32 + koff;
            uint4* fp = reinterpret_cast<uint4*>(&wfrag[kk]);
            fp[0] = *(const uint4*)(p0);
            fp[1] = *(const uint4*)(p0 + 16);
        }
    }

    // ---- Main loop: 32 pairs (two 16-pair sub-tiles) per barrier
    for (int t = 0; t < NTILE; t++) {
        unsigned short* hp = hps[t & 1];
        {   // cooperative build: 32 pairs x 128 e, 16 elems -> 2 b128 stores
            const int row  = tid >> 3;         // 0..31
            const int col0 = (tid & 7) * 16;   // 0,16,...,112
            const int p    = t * 32 + row;
            uint4 pk0 = {0u, 0u, 0u, 0u};
            uint4 pk1 = {0u, 0u, 0u, 0u};
            if (p < NPAIR) {
                const int pr = pairs[p];
                const float* xi = xs + (pr >> 8) * EDIM;
                const float* xj = xs + (pr & 255) * EDIM;
                unsigned int* w0 = (unsigned int*)&pk0;
                unsigned int* w1 = (unsigned int*)&pk1;
                #pragma unroll
                for (int c = 0; c < 4; c++) {
                    w0[c] = pack2_bf16(xi[col0 + 2*c]     * xj[col0 + 2*c],
                                       xi[col0 + 2*c + 1] * xj[col0 + 2*c + 1]);
                    w1[c] = pack2_bf16(xi[col0 + 8 + 2*c]     * xj[col0 + 8 + 2*c],
                                       xi[col0 + 8 + 2*c + 1] * xj[col0 + 8 + 2*c + 1]);
                }
            }
            *(uint4*)(hp + row * EDIM + col0)     = pk0;
            *(uint4*)(hp + row * EDIM + col0 + 8) = pk1;
        }
        __syncthreads();

        // Sub-tile A: pairs [t*32, t*32+16)
        v8f caccA = {};
        {
            const unsigned short* hrow = hp + nl * EDIM;
            #pragma unroll
            for (int kk = 0; kk < 4; kk++) {
                v16bf hfrag;
                const unsigned short* p0 = hrow + kk * 32 + koff;
                uint4* fp = reinterpret_cast<uint4*>(&hfrag);
                fp[0] = *(const uint4*)(p0);
                fp[1] = *(const uint4*)(p0 + 16);
                caccA = __builtin_amdgcn_wmma_f32_16x16x32_bf16(
                    false, wfrag[kk], false, hfrag, (short)0, caccA, false, false);
            }
        }
        // Sub-tile B: pairs [t*32+16, t*32+32)
        v8f caccB = {};
        {
            const unsigned short* hrow = hp + (nl + 16) * EDIM;
            #pragma unroll
            for (int kk = 0; kk < 4; kk++) {
                v16bf hfrag;
                const unsigned short* p0 = hrow + kk * 32 + koff;
                uint4* fp = reinterpret_cast<uint4*>(&hfrag);
                fp[0] = *(const uint4*)(p0);
                fp[1] = *(const uint4*)(p0 + 16);
                caccB = __builtin_amdgcn_wmma_f32_16x16x32_bf16(
                    false, wfrag[kk], false, hfrag, (short)0, caccB, false, false);
            }
        }

        // Epilogue: per-lane register reduction over a, xor-16 merges a-halves,
        // then ALL 32 lanes issue one conflict-free ds_add_f32 (lanes 0-15 own
        // sub-tile A columns, lanes 16-31 own sub-tile B columns). No exec mask.
        float part0 = 0.0f, part1 = 0.0f;
        #pragma unroll
        for (int r = 0; r < 8; r++) {
            part0 += fmaxf(caccA[r] + wbr[r], 0.0f) * hwr[r];
            part1 += fmaxf(caccB[r] + wbr[r], 0.0f) * hwr[r];
        }
        part0 += __shfl_xor(part0, 16, 32);
        part1 += __shfl_xor(part1, 16, 32);
        const float val = (lane < 16) ? part0 : part1;
        atomicAdd(&scores[t * 32 + ph + nl], val);
    }
    __syncthreads();

    // ---- Softmax over the 780 pair scores (block-wide)
    float mx = -3.4e38f;
    for (int p = tid; p < NPAIR; p += BLOCK) mx = fmaxf(mx, scores[p]);
    red[tid] = mx;
    __syncthreads();
    for (int s = BLOCK / 2; s > 0; s >>= 1) {
        if (tid < s) red[tid] = fmaxf(red[tid], red[tid + s]);
        __syncthreads();
    }
    mx = red[0];
    __syncthreads();

    float sm = 0.0f;
    for (int p = tid; p < NPAIR; p += BLOCK) {
        float e = __expf(scores[p] - mx);
        scores[p] = e;
        sm += e;
    }
    red[tid] = sm;
    __syncthreads();
    for (int s = BLOCK / 2; s > 0; s >>= 1) {
        if (tid < s) red[tid] += red[tid + s];
        __syncthreads();
    }
    const float inv = 1.0f / red[0];
    __syncthreads();

    // ---- afm[e] = sum_p attn[p] * x_i[e] * x_j[e]  (hp recomputed f32 from LDS x)
    if (tid < EDIM) afm_s[tid] = 0.0f;
    __syncthreads();
    {
        const int e    = tid & (EDIM - 1);
        const int half = tid >> 7;   // 0 or 1
        float acc = 0.0f;
        for (int p = half; p < NPAIR; p += 2) {
            const int pr = pairs[p];
            acc += scores[p] * xs[(pr >> 8) * EDIM + e] * xs[(pr & 255) * EDIM + e];
        }
        atomicAdd(&afm_s[e], acc * inv);
    }
    __syncthreads();

    // ---- out[b] = afm . p_w + p_b  (wave 0)
    if (tid < 32) {
        float s = 0.0f;
        #pragma unroll
        for (int c = 0; c < 4; c++)
            s += afm_s[tid + 32 * c] * p_w[tid + 32 * c];
        s += __shfl_xor(s, 1, 32);
        s += __shfl_xor(s, 2, 32);
        s += __shfl_xor(s, 4, 32);
        s += __shfl_xor(s, 8, 32);
        s += __shfl_xor(s, 16, 32);
        if (tid == 0) out[b] = s + p_b[0];
    }
}

extern "C" void kernel_launch(void* const* d_in, const int* in_sizes, int n_in,
                              void* d_out, int out_size, void* d_ws, size_t ws_size,
                              hipStream_t stream) {
    (void)n_in; (void)d_ws; (void)ws_size; (void)out_size;
    const float* x   = (const float*)d_in[0];
    const float* w_w = (const float*)d_in[1];
    const float* w_b = (const float*)d_in[2];
    const float* h_w = (const float*)d_in[3];
    const float* h_b = (const float*)d_in[4];
    const float* p_w = (const float*)d_in[5];
    const float* p_b = (const float*)d_in[6];
    float* out = (float*)d_out;

    const int B = in_sizes[0] / (NFIELD * EDIM);   // 1024
    afm_fused_kernel<<<B, BLOCK, 0, stream>>>(x, w_w, w_b, h_w, h_b, p_w, p_b, out);
}
